// NonLocalAttention_38671885533891
// MI455X (gfx1250) — matrix operand order repaired and checked by reading, not compiled, not run on hardware
//
#include <hip/hip_runtime.h>
#include <hip/hip_bf16.h>
#include <stdint.h>

#define B_   16
#define CIN  256
#define CA_  32
#define CH_  128
#define N_   4096
#define M_   1024
#define OCT  192      // 32 theta + 32 phi + 128 g channels in projF

typedef __attribute__((ext_vector_type(16))) __bf16 v16bf;
typedef __attribute__((ext_vector_type(8)))  float  v8f;

union Frag { v16bf bf; uint4 q[2]; };
union Pack8 { uint4 q; unsigned short s[8]; };

__device__ __forceinline__ unsigned short f2bf(float f) {
  union { float f; unsigned u; } c; c.f = f;
  unsigned u = c.u + 0x7FFFu + ((c.u >> 16) & 1u);
  return (unsigned short)(u >> 16);
}
__device__ __forceinline__ float bf2f(unsigned short h) {
  union { unsigned u; float f; } c; c.u = ((unsigned)h) << 16;
  return c.f;
}

// ---------------------------------------------------------------------------
// Kernel 1: fused 1x1-conv projections as one WMMA GEMM into a unified
// projF (B, N, 192) bf16 buffer: ch 0-31 = theta, 32-63 = phiF, 64-191 = gF.
// Block = 256 thr (8 waves), one batch + 2-row strip (128 px).
// Each wave owns one 16-row A (weight) tile: A fragments hoisted to regs,
// B (x) tiles streamed from LDS.  Packed b128 epilogue stores.
// ---------------------------------------------------------------------------
__global__ __launch_bounds__(256) void proj_kernel(
    const float* __restrict__ x, const float* __restrict__ wt,
    const float* __restrict__ wp, const float* __restrict__ wg,
    unsigned short* __restrict__ projF) {
  __shared__ __align__(16) unsigned short sW[32 * 256];   // 16 KB weight chunk
  __shared__ __align__(16) unsigned short sX[128 * 264];  // 66 KB x^T, padded

  const int tid   = threadIdx.x;
  const int b     = blockIdx.x >> 5;
  const int strip = blockIdx.x & 31;
  const int nbase = strip * 128;

  // Stage x strip transposed: sX[px][c] in bf16 (coalesced global reads).
  {
    const int px = tid & 127;
    for (int c = tid >> 7; c < CIN; c += 2) {
      float v = x[((size_t)(b * CIN + c)) * N_ + nbase + px];
      sX[px * 264 + c] = f2bf(v);
    }
  }

  const int wave   = tid >> 5;
  const int lane   = tid & 31;
  const int lane16 = lane & 15;
  const int hiA = (lane < 16) ? 0 : 16;   // A-matrix K base
  const int hiB = (lane < 16) ? 0 : 16;   // B-matrix K base
  const int hiD = (lane < 16) ? 0 : 8;    // D-matrix row base
  const int octile = wave >> 2;           // waves 0-3 -> octile 0, 4-7 -> 1
  const int arow   = octile * 16 + lane16;

  for (int occ = 0; occ < 6; ++occ) {     // 6 chunks of 32 output channels
    __syncthreads();
    {   // stage 32 weight rows (theta rows 0-31, phi 32-63, g 64-191)
      const int col = tid;
      for (int i = 0; i < 32; ++i) {
        int rg = occ * 32 + i;
        const float* src = (rg < 32) ? (wt + rg * CIN)
                         : (rg < 64) ? (wp + (rg - 32) * CIN)
                                     : (wg + (rg - 64) * CIN);
        sW[i * 256 + col] = f2bf(src[col]);
      }
    }
    __syncthreads();

    // Hoist this wave's A fragments (all K) to registers once per chunk.
    Frag aF[8];
#pragma unroll
    for (int kc = 0; kc < 8; ++kc) {
      aF[kc].q[0] = *(const uint4*)&sW[arow * 256 + kc * 32 + hiA];
      aF[kc].q[1] = *(const uint4*)&sW[arow * 256 + kc * 32 + hiA + 8];
    }

    // Stream 2 px-tiles per wave against the resident A.
    for (int tt = 0; tt < 2; ++tt) {
      const int pxtile = (wave & 3) * 2 + tt;
      const int bpx    = pxtile * 16 + lane16;
      v8f acc = {0.f,0.f,0.f,0.f,0.f,0.f,0.f,0.f};
#pragma unroll
      for (int kc = 0; kc < 8; ++kc) {    // K = 256 = 8 * 32
        Frag bm;
        bm.q[0] = *(const uint4*)&sX[bpx * 264 + kc * 32 + hiB];
        bm.q[1] = *(const uint4*)&sX[bpx * 264 + kc * 32 + hiB + 8];
        acc = __builtin_amdgcn_wmma_f32_16x16x32_bf16(
            false, aF[kc].bf, false, bm.bf, (short)0, acc, false, false);
      }
      // 8 results per lane are 8 consecutive channels -> one b128 store
      const int n        = nbase + bpx;
      const int tilebase = occ * 32 + octile * 16;
      Pack8 pk;
#pragma unroll
      for (int r = 0; r < 8; ++r) pk.s[r] = f2bf(acc[r]);
      *(uint4*)&projF[((size_t)b * N_ + n) * OCT + tilebase + hiD] = pk.q;
    }
  }
}

// ---------------------------------------------------------------------------
// Kernel 2: 2x2 max pool of phi/g (read from projF) into WMMA layouts.
// phi_p: (B, M, CA)  key-major;  g_p: (B, CH, M)  contraction-contiguous.
// Block-uniform split: first blocks do phi, rest do g.
// ---------------------------------------------------------------------------
__global__ __launch_bounds__(256) void pool_kernel(
    const unsigned short* __restrict__ projF,
    unsigned short* __restrict__ phi_p, unsigned short* __restrict__ g_p) {
  const int PHI_TOT = B_ * M_ * CA_;              // 524288 (2048 blocks)
  int idx = blockIdx.x * 256 + threadIdx.x;
  if (idx < PHI_TOT) {
    int ch = idx & 31;
    int m  = (idx >> 5) & (M_ - 1);
    int b  = idx >> 15;
    int pr = m >> 5, pc = m & 31;
    int n0 = pr * 128 + pc * 2;
    float mv = fmaxf(
        fmaxf(bf2f(projF[((size_t)b * N_ + n0)      * OCT + 32 + ch]),
              bf2f(projF[((size_t)b * N_ + n0 + 1)  * OCT + 32 + ch])),
        fmaxf(bf2f(projF[((size_t)b * N_ + n0 + 64) * OCT + 32 + ch]),
              bf2f(projF[((size_t)b * N_ + n0 + 65) * OCT + 32 + ch])));
    phi_p[((size_t)b * M_ + m) * CA_ + ch] = f2bf(mv);
  } else {
    int j = idx - PHI_TOT;                        // over B*M*CH
    int m = j & (M_ - 1);
    int c = (j >> 10) & 127;
    int b = j >> 17;
    int pr = m >> 5, pc = m & 31;
    int n0 = pr * 128 + pc * 2;
    float mv = fmaxf(
        fmaxf(bf2f(projF[((size_t)b * N_ + n0)      * OCT + 64 + c]),
              bf2f(projF[((size_t)b * N_ + n0 + 1)  * OCT + 64 + c])),
        fmaxf(bf2f(projF[((size_t)b * N_ + n0 + 64) * OCT + 64 + c]),
              bf2f(projF[((size_t)b * N_ + n0 + 65) * OCT + 64 + c])));
    g_p[((size_t)b * CH_ + c) * M_ + m] = f2bf(mv);
  }
}

// ---------------------------------------------------------------------------
// Kernel 3: flash attention.  Block = 256 thr (8 waves), one batch + 128
// queries (16 per wave) -> phi/g tile fetches amortized over 2x more queries.
// BK=32 keys/iter: 2 score WMMAs (K=CA=32), online softmax via shfl row
// reductions in the D lane layout, P restaged through per-wave LDS into
// A layout, 8 PV WMMAs (CH=128).  Output staged in LDS -> b128 stores.
// ---------------------------------------------------------------------------
__global__ __launch_bounds__(256) void attn_kernel(
    const unsigned short* __restrict__ projF, const unsigned short* __restrict__ phi_p,
    const unsigned short* __restrict__ g_p, unsigned short* __restrict__ o) {
  __shared__ __align__(16) unsigned short sPhi[32 * 32];     // [m_local][ca]
  __shared__ __align__(16) unsigned short sG[128 * 32];      // [c][m_local]
  __shared__ __align__(16) unsigned short sP[8][16 * 32];    // per-wave [q][k]
  __shared__ __align__(16) unsigned short sO[128 * 136];     // [q][c] padded

  const int tid    = threadIdx.x;
  const int b      = blockIdx.x >> 5;
  const int qb     = blockIdx.x & 31;
  const int wave   = tid >> 5;
  const int lane   = tid & 31;
  const int lane16 = lane & 15;
  const int hiA = (lane < 16) ? 0 : 16;
  const int hiB = (lane < 16) ? 0 : 16;
  const int hiD = (lane < 16) ? 0 : 8;
  const int q0  = qb * 128 + wave * 16;

  Frag aTh;   // theta A-tile from projF channels 0-31 (row stride 192)
  {
    const unsigned short* tp = &projF[((size_t)b * N_ + q0 + lane16) * OCT];
    aTh.q[0] = *(const uint4*)&tp[hiA];
    aTh.q[1] = *(const uint4*)&tp[hiA + 8];
  }

  const v8f vzero = {0.f,0.f,0.f,0.f,0.f,0.f,0.f,0.f};
  v8f oacc[8];
  float rmax[8], rsum[8];
#pragma unroll
  for (int t = 0; t < 8; ++t) oacc[t] = vzero;
#pragma unroll
  for (int r = 0; r < 8; ++r) { rmax[r] = -3.0e38f; rsum[r] = 0.f; }

  for (int m0 = 0; m0 < M_; m0 += 32) {
    __syncthreads();
    if (tid < 128) {    // phi tile: 32 rows x 4 chunks = 128 x 16B
      int ml = tid >> 2, part = tid & 3;
      const unsigned short* pp = &phi_p[((size_t)b * M_ + m0 + ml) * CA_];
      *(uint4*)&sPhi[ml * 32 + part * 8] = *(const uint4*)&pp[part * 8];
      if (m0 + 32 < M_) __builtin_prefetch(pp + 32 * CA_, 0, 1);
    } else {            // g tile: 128 rows x 4 chunks
      int c = tid - 128;
      const unsigned short* gp = &g_p[((size_t)(b * CH_ + c)) * M_ + m0];
#pragma unroll
      for (int j = 0; j < 4; ++j)
        *(uint4*)&sG[c * 32 + j * 8] = *(const uint4*)&gp[j * 8];
      if (m0 + 32 < M_) __builtin_prefetch(gp + 32, 0, 1);
    }
    __syncthreads();

    // scores: S = theta(16x32) @ phi^T(32x16), two key sub-tiles
    v8f s0, s1;
    {
      Frag b0, b1;
      b0.q[0] = *(const uint4*)&sPhi[lane16 * 32 + hiB];
      b0.q[1] = *(const uint4*)&sPhi[lane16 * 32 + hiB + 8];
      b1.q[0] = *(const uint4*)&sPhi[(16 + lane16) * 32 + hiB];
      b1.q[1] = *(const uint4*)&sPhi[(16 + lane16) * 32 + hiB + 8];
      s0 = __builtin_amdgcn_wmma_f32_16x16x32_bf16(
          false, aTh.bf, false, b0.bf, (short)0, vzero, false, false);
      s1 = __builtin_amdgcn_wmma_f32_16x16x32_bf16(
          false, aTh.bf, false, b1.bf, (short)0, vzero, false, false);
    }

    // online softmax per query row (row = r + hiD, keys across 16 lanes)
    float scale[8];
#pragma unroll
    for (int r = 0; r < 8; ++r) {
      float v0 = s0[r], v1 = s1[r];
      float mx = fmaxf(v0, v1);
      mx = fmaxf(mx, __shfl_xor(mx, 1, 16));
      mx = fmaxf(mx, __shfl_xor(mx, 2, 16));
      mx = fmaxf(mx, __shfl_xor(mx, 4, 16));
      mx = fmaxf(mx, __shfl_xor(mx, 8, 16));
      float nm = fmaxf(rmax[r], mx);
      float sc = __expf(rmax[r] - nm);
      float p0 = __expf(v0 - nm);
      float p1 = __expf(v1 - nm);
      float ps = p0 + p1;
      ps += __shfl_xor(ps, 1, 16);
      ps += __shfl_xor(ps, 2, 16);
      ps += __shfl_xor(ps, 4, 16);
      ps += __shfl_xor(ps, 8, 16);
      rmax[r]  = nm;
      rsum[r]  = rsum[r] * sc + ps;
      scale[r] = sc;
      int qrow = r + hiD;   // D layout -> A layout transpose via LDS
      sP[wave][qrow * 32 + lane16]      = f2bf(p0);
      sP[wave][qrow * 32 + 16 + lane16] = f2bf(p1);
    }
#pragma unroll
    for (int t = 0; t < 8; ++t)
#pragma unroll
      for (int r = 0; r < 8; ++r) oacc[t][r] *= scale[r];

    // PV: O += P(16x32) @ g^T(32x16) per channel tile
    Frag aP;
    aP.q[0] = *(const uint4*)&sP[wave][lane16 * 32 + hiA];
    aP.q[1] = *(const uint4*)&sP[wave][lane16 * 32 + hiA + 8];
#pragma unroll
    for (int t = 0; t < 8; ++t) {
      Frag bg;
      int c = t * 16 + lane16;
      bg.q[0] = *(const uint4*)&sG[c * 32 + hiB];
      bg.q[1] = *(const uint4*)&sG[c * 32 + hiB + 8];
      oacc[t] = __builtin_amdgcn_wmma_f32_16x16x32_bf16(
          false, aP.bf, false, bg.bf, (short)0, oacc[t], false, false);
    }
  }

  // finalize: divide by row sums, stage in LDS, coalesced b128 stores
#pragma unroll
  for (int r = 0; r < 8; ++r) rsum[r] = 1.0f / rsum[r];
#pragma unroll
  for (int t = 0; t < 8; ++t) {
    int c = t * 16 + lane16;
#pragma unroll
    for (int r = 0; r < 8; ++r)
      sO[(wave * 16 + r + hiD) * 136 + c] = f2bf(oacc[t][r] * rsum[r]);
  }
  __syncthreads();
  {
    const size_t obase = ((size_t)b * N_ + qb * 128) * CH_;
#pragma unroll
    for (int j = 0; j < 8; ++j) {
      int chunk = j * 256 + tid;            // 2048 x 16B chunks
      int q = chunk >> 4, part = chunk & 15;
      *(uint4*)&o[obase + (size_t)q * CH_ + part * 8] =
          *(const uint4*)&sO[q * 136 + part * 8];
    }
  }
}

// ---------------------------------------------------------------------------
// Kernel 4: out = gamma * (W_o @ o) + x, WMMA with residual epilogue.
// Block = 256 thr (8 waves), one batch + 256 pixels.  Each wave owns 2
// co-tiles; per co-tile the A (W_o) fragments are hoisted to registers and
// 16 px-tiles streamed against them (B straight from global o rows).
// ---------------------------------------------------------------------------
__global__ __launch_bounds__(256) void outproj_kernel(
    const unsigned short* __restrict__ o, const float* __restrict__ wo,
    const float* __restrict__ x, const float* __restrict__ gammap,
    float* __restrict__ out) {
  __shared__ __align__(16) unsigned short sWo[256 * 128];   // 64 KB

  const int tid = threadIdx.x;
  const int b   = blockIdx.x >> 4;
  const int pxg = blockIdx.x & 15;
  for (int i = 0; i < 128; ++i) {
    int idx = i * 256 + tid;
    sWo[idx] = f2bf(wo[idx]);
  }
  __syncthreads();

  const float gamma = gammap[0];
  const int wave   = tid >> 5;
  const int lane   = tid & 31;
  const int lane16 = lane & 15;
  const int hiA = (lane < 16) ? 0 : 16;
  const int hiB = (lane < 16) ? 0 : 16;
  const int hiD = (lane < 16) ? 0 : 8;

  for (int cc = 0; cc < 2; ++cc) {
    const int cotile = wave * 2 + cc;              // 16 co-tiles
    const int arow   = cotile * 16 + lane16;
    Frag aF[4];
#pragma unroll
    for (int kc = 0; kc < 4; ++kc) {               // K = CH = 128
      aF[kc].q[0] = *(const uint4*)&sWo[arow * 128 + kc * 32 + hiA];
      aF[kc].q[1] = *(const uint4*)&sWo[arow * 128 + kc * 32 + hiA + 8];
    }
    for (int pxtile = 0; pxtile < 16; ++pxtile) {
      const int px = pxg * 256 + pxtile * 16 + lane16;
      const unsigned short* op = &o[((size_t)b * N_ + px) * CH_];
      v8f acc = {0.f,0.f,0.f,0.f,0.f,0.f,0.f,0.f};
#pragma unroll
      for (int kc = 0; kc < 4; ++kc) {
        Frag bm;
        bm.q[0] = *(const uint4*)&op[kc * 32 + hiB];
        bm.q[1] = *(const uint4*)&op[kc * 32 + hiB + 8];
        acc = __builtin_amdgcn_wmma_f32_16x16x32_bf16(
            false, aF[kc].bf, false, bm.bf, (short)0, acc, false, false);
      }
#pragma unroll
      for (int r = 0; r < 8; ++r) {
        int co = cotile * 16 + r + hiD;
        size_t addr = ((size_t)b * CIN + co) * N_ + px;
        out[addr] = gamma * acc[r] + x[addr];
      }
    }
  }
}

// ---------------------------------------------------------------------------
extern "C" void kernel_launch(void* const* d_in, const int* in_sizes, int n_in,
                              void* d_out, int out_size, void* d_ws, size_t ws_size,
                              hipStream_t stream) {
  (void)in_sizes; (void)n_in; (void)out_size; (void)ws_size;
  const float* x     = (const float*)d_in[0];
  const float* wt    = (const float*)d_in[1];
  const float* wp    = (const float*)d_in[2];
  const float* wg    = (const float*)d_in[3];
  const float* wo    = (const float*)d_in[4];
  const float* gamma = (const float*)d_in[5];
  float* out = (float*)d_out;

  char* ws = (char*)d_ws;
  unsigned short* projF = (unsigned short*)(ws);             // 24 MB (B,N,192)
  unsigned short* phi_p = (unsigned short*)(ws + 25165824);  //  1 MB (B,M,CA)
  unsigned short* g_p   = (unsigned short*)(ws + 26214400);  //  4 MB (B,CH,M)
  unsigned short* o     = (unsigned short*)(ws + 30408704);  // 16 MB (B,N,CH)

  proj_kernel<<<B_ * 32, 256, 0, stream>>>(x, wt, wp, wg, projF);
  pool_kernel<<<(B_ * M_ * 160 + 255) / 256, 256, 0, stream>>>(projF, phi_p, g_p);
  attn_kernel<<<B_ * 32, 256, 0, stream>>>(projF, phi_p, g_p, o);
  outproj_kernel<<<B_ * 16, 256, 0, stream>>>(o, wo, x, gamma, out);
}
